// PatchMerged_61881888801151
// MI455X (gfx1250) — compile-verified, matrix operand order checked by reading, not run on hardware
//
#include <hip/hip_runtime.h>
#include <hip/hip_bf16.h>

typedef __attribute__((ext_vector_type(16))) _Float16 v16h;
typedef __attribute__((ext_vector_type(8)))  float    v8f;

#define Bm  16
#define Cc  96
#define Hh  256
#define Ww  256
#define C4  384
#define C2  192
#define HO  128
#define WO  128
#define TPB 768          // 24 waves: wave = (nt[0..11], mhalf[0..1])

// LDS layout: only the Y tile + per-row stats live in LDS now (~111 KB),
// so 2 WGs fit per WGP (320 KB) and DWT-streaming overlaps GEMM across WGs.
#define YS  400          // halves per Y row (800B, 16B aligned)
#define OFF_Y   0
#define OFF_MU  (HO * YS * 2)            // 102400
#define OFF_RS  (OFF_MU + HO * 4)        // 102912
#define OFF_U   (OFF_RS + HO * 4)        // 103424
#define OFF_V   (OFF_U + C2 * 4)         // 104192
#define OFF_R1  (OFF_V + C2 * 4)         // 104960  (6 x 128 floats)
#define OFF_R2  (OFF_R1 + 6 * HO * 4)    // 108032
#define SMEM_BYTES (OFF_R2 + 6 * HO * 4) // 111104

// ---------------------------------------------------------------------------
// Prep: fold norm_w into W, transpose to f16 W'^T[192][384], colsums u,v.
// LayerNorm folds into GEMM epilogue: out = rstd*(y@W') - rstd*mu*u + v
// ---------------------------------------------------------------------------
__global__ void prep_kernel(const float* __restrict__ w_red,
                            const float* __restrict__ g,
                            const float* __restrict__ nb,
                            _Float16* __restrict__ wt,
                            float* __restrict__ u,
                            float* __restrict__ v) {
    int n = blockIdx.x * blockDim.x + threadIdx.x;
    if (n >= C2) return;
    float su = 0.f, sv = 0.f;
    for (int k = 0; k < C4; ++k) {
        float wv = w_red[(size_t)k * C2 + n];
        float gw = g[k] * wv;
        wt[(size_t)n * C4 + k] = (_Float16)gw;
        su += gw;
        sv += nb[k] * wv;
    }
    u[n] = su;
    v[n] = sv;
}

// ---------------------------------------------------------------------------
// Fused DWT + LN-stats + WMMA GEMM. One WG per (batch, output row h).
// M-tile = 128 rows, N = 192 (12 tiles, one per wave-pair), K = 384 (12 steps).
// B fragments live in VGPRs (loaded from L2-resident W'^T before the barrier).
// ---------------------------------------------------------------------------
__global__ __launch_bounds__(TPB) void fused_kernel(
        const float* __restrict__ x,
        const _Float16* __restrict__ wt_g,
        const float* __restrict__ u_g,
        const float* __restrict__ v_g,
        float* __restrict__ out) {
    extern __shared__ char smem[];
    _Float16* Yl  = (_Float16*)(smem + OFF_Y);
    float*    muA = (float*)(smem + OFF_MU);
    float*    rsA = (float*)(smem + OFF_RS);
    float*    uA  = (float*)(smem + OFF_U);
    float*    vA  = (float*)(smem + OFF_V);
    float*    r1  = (float*)(smem + OFF_R1);
    float*    r2  = (float*)(smem + OFF_R2);

    const int t = threadIdx.x;
    const int h = blockIdx.x;   // 0..127
    const int b = blockIdx.y;   // 0..15

    const int wave  = t >> 5;
    const int lane  = t & 31;
    const int nt    = wave >> 1;       // N-tile (0..11)
    const int mhalf = wave & 1;        // which 4 M-tiles (0..1)
    const int ncol  = lane & 15;
    const int koff  = (lane >> 4) * 8; // lanes 0-15: K{0-7,16-23}; 16-31: {8-15,24-31}

    union Frag { v16h v; uint4 q[2]; };

    // ---- B fragments: 12 K-steps for this wave's N column, straight from L2.
    // Issued before the DWT so the loads overlap the x streaming below.
    Frag Bf[12];
    {
        const char* wp = (const char*)(wt_g + (size_t)(nt * 16 + ncol) * C4 + koff);
        #pragma unroll
        for (int kt = 0; kt < 12; ++kt) {
            Bf[kt].q[0] = *(const uint4*)(wp + kt * 64);
            Bf[kt].q[1] = *(const uint4*)(wp + kt * 64 + 32);
        }
    }
    if (t < C2) { uA[t] = u_g[t]; vA[t] = v_g[t]; }

    // ---- Haar DWT for 128 positions; 6 threads per position (16 channels each)
    {
        const int w    = t & (WO - 1);
        const int part = t >> 7;                 // 0..5
        float s1 = 0.f, s2 = 0.f;
        const float* xb = x + ((size_t)(b * Cc + part * 16) * Hh + 2 * h) * Ww + 2 * w;
        _Float16* yrow = Yl + (size_t)w * YS;
        #pragma unroll 4
        for (int ci = 0; ci < 16; ++ci) {
            const float* p = xb + (size_t)ci * (Hh * Ww);
            float2 q0 = *(const float2*)p;          // x[2h, 2w..2w+1]
            float2 q1 = *(const float2*)(p + Ww);   // x[2h+1, 2w..2w+1]
            float a = q0.x, bb = q0.y, cc = q1.x, dd = q1.y;
            float ll = (a + bb + cc + dd) * 0.5f;
            float lh = (a + bb - cc - dd) * 0.5f;
            float hl = (a - bb + cc - dd) * 0.5f;
            float hh = (a - bb - cc + dd) * 0.5f;
            int c4 = part * 16 + ci;
            yrow[c4]       = (_Float16)ll;
            yrow[96  + c4] = (_Float16)lh;
            yrow[192 + c4] = (_Float16)hl;
            yrow[288 + c4] = (_Float16)hh;
            s1 += ll + lh + hl + hh;
            s2 += ll * ll + lh * lh + hl * hl + hh * hh;
        }
        r1[part * WO + w] = s1;
        r2[part * WO + w] = s2;
    }
    __syncthreads();

    if (t < WO) {
        float s1 = 0.f, s2 = 0.f;
        #pragma unroll
        for (int p = 0; p < 6; ++p) { s1 += r1[p * WO + t]; s2 += r2[p * WO + t]; }
        float mu  = s1 * (1.0f / C4);
        float var = s2 * (1.0f / C4) - mu * mu;
        muA[t] = mu;
        rsA[t] = rsqrtf(var + 1e-5f);
    }
    __syncthreads();

    // ---- WMMA GEMM: 4 M-tiles per wave, one accumulator live at a time.
    const int n  = nt * 16 + ncol;
    const float un = uA[n], vn = vA[n];
    const int rhi = (lane >> 4) * 8;    // C layout: vgpr j -> M=j (lo), j+8 (hi)
    const size_t outbase = ((size_t)b * (HO * WO) + (size_t)h * WO) * C2;

    #pragma unroll
    for (int mt = 0; mt < 4; ++mt) {
        const int m0 = (mhalf * 4 + mt) * 16;
        const char* Abase = smem + OFF_Y + (m0 + (lane & 15)) * (YS * 2) + koff * 2;

        v8f acc = {0.f, 0.f, 0.f, 0.f, 0.f, 0.f, 0.f, 0.f};
        #pragma unroll
        for (int kt = 0; kt < 12; ++kt) {
            Frag af;
            af.q[0] = *(const uint4*)(Abase + kt * 64);
            af.q[1] = *(const uint4*)(Abase + kt * 64 + 32);
            acc = __builtin_amdgcn_wmma_f32_16x16x32_f16(
                false, af.v, false, Bf[kt].v, (short)0, acc, false, false);
        }

        // Epilogue: out = rstd*S - rstd*mu*u[n] + v[n]
        #pragma unroll
        for (int j = 0; j < 8; ++j) {
            int w = m0 + rhi + j;
            float mu = muA[w], rs = rsA[w];
            out[outbase + (size_t)w * C2 + n] = rs * acc[j] - rs * mu * un + vn;
        }
    }
}

// ---------------------------------------------------------------------------
extern "C" void kernel_launch(void* const* d_in, const int* in_sizes, int n_in,
                              void* d_out, int out_size, void* d_ws, size_t ws_size,
                              hipStream_t stream) {
    const float* x      = (const float*)d_in[0];
    const float* norm_w = (const float*)d_in[1];
    const float* norm_b = (const float*)d_in[2];
    const float* w_red  = (const float*)d_in[3];
    float* out = (float*)d_out;

    _Float16* wt = (_Float16*)d_ws;                       // 192*384 halves
    float* u = (float*)((char*)d_ws + (size_t)C2 * C4 * 2);
    float* v = u + C2;

    prep_kernel<<<dim3((C2 + 63) / 64), dim3(64), 0, stream>>>(
        w_red, norm_w, norm_b, wt, u, v);

    (void)hipFuncSetAttribute((const void*)fused_kernel,
                              hipFuncAttributeMaxDynamicSharedMemorySize,
                              SMEM_BYTES);
    fused_kernel<<<dim3(HO, Bm), dim3(TPB), SMEM_BYTES, stream>>>(
        x, wt, u, v, out);
}
